// MlaAttention_2216203124742
// MI455X (gfx1250) — matrix operand order repaired and checked by reading, not compiled, not run on hardware
//
#include <hip/hip_runtime.h>
#include <hip/hip_bf16.h>

#define HIDDEN 4096
#define Q_LORA 1536
#define KV_LORA 512
#define NOPE 128
#define ROPE 64
#define V_DIM 128
#define NHEAD 32
#define Q_HEAD 192           // NOPE + ROPE
#define NA 2112              // Q_LORA + KV_LORA + ROPE
#define EPS 1e-6f

typedef __attribute__((ext_vector_type(16))) __bf16 bf16x16;
typedef __attribute__((ext_vector_type(8)))  __bf16 bf16x8;
typedef __attribute__((ext_vector_type(8)))  float  f32x8;
typedef __attribute__((ext_vector_type(4)))  float  f32x4;
typedef int v4i_ __attribute__((vector_size(16)));

// --- gfx1250 async global->LDS path (guarded; sync fallback) ---------------
#if defined(__HIP_DEVICE_COMPILE__) &&                                        \
    __has_builtin(__builtin_amdgcn_global_load_async_to_lds_b128) &&          \
    __has_builtin(__builtin_amdgcn_s_wait_asynccnt)
#define HAS_ASYNC_LDS 1
#else
#define HAS_ASYNC_LDS 0
#endif

__device__ __forceinline__ bf16x8 ld8(const __bf16* p) { return *(const bf16x8*)p; }

// copy 16 bytes global -> LDS (async when available)
__device__ __forceinline__ void cp16(void* lds, const void* g) {
#if HAS_ASYNC_LDS
    __builtin_amdgcn_global_load_async_to_lds_b128(
        (__attribute__((address_space(1))) v4i_*)g,
        (__attribute__((address_space(3))) v4i_*)lds, 0, 0);
#else
    *(bf16x8*)lds = *(const bf16x8*)g;
#endif
}

__device__ __forceinline__ void async_wait0() {
#if HAS_ASYNC_LDS
    __builtin_amdgcn_s_wait_asynccnt(0);
#endif
}

__device__ __forceinline__ bf16x16 cat16(bf16x8 lo, bf16x8 hi) {
    bf16x16 r;
#pragma unroll
    for (int i = 0; i < 8; ++i) { r[i] = lo[i]; r[8 + i] = hi[i]; }
    return r;
}

__device__ __forceinline__ f32x8 wmma_bf16(bf16x16 a, bf16x16 b, f32x8 c) {
    return __builtin_amdgcn_wmma_f32_16x16x32_bf16(false, a, false, b, (short)0, c,
                                                   false, false);
}

__device__ __forceinline__ void lds_raw_fence() {
#if defined(__HIP_DEVICE_COMPILE__) && __has_builtin(__builtin_amdgcn_s_wait_dscnt)
    __builtin_amdgcn_s_wait_dscnt(0);
#else
    asm volatile("" ::: "memory");
#endif
}

__device__ __forceinline__ f32x8 zero8() {
    return (f32x8){0.f, 0.f, 0.f, 0.f, 0.f, 0.f, 0.f, 0.f};
}

// ---------------------------------------------------------------------------
// Tiled GEMM: C[M,N] = A[M,K] @ B[K,N] (+bias). B source f32, converted to
// bf16 while staging (transposed) into LDS. A is f32 (convert) or bf16 (async
// 16B copies). Block 256 = 8 waves; tile 128x128, BK=64, double-buffered.
// Wave = 32x64 region (2x4 WMMA tiles), 16 WMMAs per K-step.
// ---------------------------------------------------------------------------
template <bool A_F32, bool C_F32, bool HAS_BIAS>
__global__ __launch_bounds__(256) void gemm_bf16_kernel(
    const void* __restrict__ Aptr, const float* __restrict__ B,
    void* __restrict__ Cptr, const float* __restrict__ bias,
    int M, int N, int K) {
    constexpr int BM = 128, BN = 128, BK = 64;
    constexpr int LDS = BK + 8;  // 72 elems = 144B (16B multiple)
    __shared__ __bf16 As[2][BM][LDS];
    __shared__ __bf16 Bs[2][BN][LDS];  // transposed: [n][k]

    const int tid  = threadIdx.x;
    const int lane = tid & 31;
    const int wave = tid >> 5;
    const int wm   = wave >> 1;  // 0..3 -> 32-row strip
    const int wn   = wave & 1;   // 0..1 -> 64-col strip
    const int m0   = blockIdx.y * BM;
    const int n0   = blockIdx.x * BN;

    f32x8 acc[2][4];
#pragma unroll
    for (int i = 0; i < 2; ++i)
#pragma unroll
        for (int j = 0; j < 4; ++j) acc[i][j] = zero8();

    auto stage = [&](int buf, int k0) {
        // ---- A tile ----
        if (A_F32) {
            const float* A = (const float*)Aptr;
#pragma unroll
            for (int i = 0; i < 8; ++i) {
                int idx = tid + i * 256;  // 2048 float4 chunks
                int r = idx >> 4, c4 = (idx & 15) * 4;
                f32x4 v = *(const f32x4*)(A + (size_t)(m0 + r) * K + k0 + c4);
#pragma unroll
                for (int j = 0; j < 4; ++j) As[buf][r][c4 + j] = (__bf16)v[j];
            }
        } else {
            const __bf16* A = (const __bf16*)Aptr;
#pragma unroll
            for (int i = 0; i < 4; ++i) {
                int idx = tid + i * 256;  // 1024 bf16x8 chunks
                int r = idx >> 3, c8 = (idx & 7) * 8;
                cp16(&As[buf][r][c8], A + (size_t)(m0 + r) * K + k0 + c8);
            }
        }
        // ---- B tile (convert + transpose into [n][k]) ----
#pragma unroll
        for (int i = 0; i < 8; ++i) {
            int idx = tid + i * 256;  // 2048 float4 chunks
            int kr = idx >> 5, c4 = (idx & 31) * 4;
            int n = n0 + c4;
            f32x4 v = {0.f, 0.f, 0.f, 0.f};
            if (n < N) v = *(const f32x4*)(B + (size_t)(k0 + kr) * N + n);
#pragma unroll
            for (int j = 0; j < 4; ++j) Bs[buf][c4 + j][kr] = (__bf16)v[j];
        }
    };

    const int nsteps = K / BK;
    stage(0, 0);
    async_wait0();
    __syncthreads();

    for (int s = 0; s < nsteps; ++s) {
        const int cur = s & 1;
        if (s + 1 < nsteps) stage(1 - cur, (s + 1) * BK);  // prefetch next buffer

#pragma unroll
        for (int kk = 0; kk < BK; kk += 32) {
            bf16x16 af[2], bfr[4];
            int r  = wm * 32 + (lane & 15);
            int kb = kk + ((lane < 16) ? 0 : 8);
#pragma unroll
            for (int mf = 0; mf < 2; ++mf)
                af[mf] = cat16(ld8(&As[cur][r + mf * 16][kb]),
                               ld8(&As[cur][r + mf * 16][kb + 16]));
            int c   = wn * 64 + (lane & 15);
            int kb2 = kk + ((lane < 16) ? 0 : 8);
#pragma unroll
            for (int nf = 0; nf < 4; ++nf)
                bfr[nf] = cat16(ld8(&Bs[cur][c + nf * 16][kb2]),
                                ld8(&Bs[cur][c + nf * 16][kb2 + 16]));
#pragma unroll
            for (int mf = 0; mf < 2; ++mf)
#pragma unroll
                for (int nf = 0; nf < 4; ++nf)
                    acc[mf][nf] = wmma_bf16(af[mf], bfr[nf], acc[mf][nf]);
        }
        async_wait0();
        __syncthreads();
    }

    // ---- epilogue: C layout VGPR i -> row (i + 8*(lane>>4)), col lane&15 ----
    const int rhalf = 8 * (lane >> 4);
    const int cl    = lane & 15;
#pragma unroll
    for (int mf = 0; mf < 2; ++mf) {
#pragma unroll
        for (int nf = 0; nf < 4; ++nf) {
            int gc = n0 + wn * 64 + nf * 16 + cl;
            if (gc >= N) continue;
            float bv = HAS_BIAS ? bias[gc] : 0.f;
#pragma unroll
            for (int i = 0; i < 8; ++i) {
                int gr = m0 + wm * 32 + mf * 16 + i + rhalf;
                float v = acc[mf][nf][i] + bv;
                if (C_F32)
                    ((float*)Cptr)[(size_t)gr * N + gc] = v;
                else
                    ((__bf16*)Cptr)[(size_t)gr * N + gc] = (__bf16)v;
            }
        }
    }
}

// ---------------------------------------------------------------------------
// RMSNorm one row per block (width W in {1536,512}), output bf16.
// ---------------------------------------------------------------------------
template <int W>
__global__ __launch_bounds__(256) void rmsnorm_kernel(
    const float* __restrict__ in, int in_stride,
    const float* __restrict__ gamma, __bf16* __restrict__ out) {
    constexpr int PER = W / 256;
    const int row = blockIdx.x, tid = threadIdx.x;
    const float* x = in + (size_t)row * in_stride;
    float v[PER], ss = 0.f;
#pragma unroll
    for (int i = 0; i < PER; ++i) { v[i] = x[tid + i * 256]; ss += v[i] * v[i]; }
#pragma unroll
    for (int m = 16; m >= 1; m >>= 1) ss += __shfl_xor(ss, m, 32);
    __shared__ float red[8];
    if ((tid & 31) == 0) red[tid >> 5] = ss;
    __syncthreads();
    float tot = 0.f;
#pragma unroll
    for (int j = 0; j < 8; ++j) tot += red[j];
    float r = rsqrtf(tot / (float)W + EPS);
#pragma unroll
    for (int i = 0; i < PER; ++i) {
        int c = tid + i * 256;
        out[(size_t)row * W + c] = (__bf16)(v[i] * r * gamma[c]);
    }
}

__global__ __launch_bounds__(256) void kpe_cvt_kernel(
    const float* __restrict__ fused, __bf16* __restrict__ kpe, int total) {
    int i = blockIdx.x * 256 + threadIdx.x;
    if (i < total) {
        int r = i >> 6, c = i & 63;
        kpe[i] = (__bf16)fused[(size_t)r * NA + (Q_LORA + KV_LORA) + c];
    }
}

// ---------------------------------------------------------------------------
// Flash attention. Grid: (S/128, H). Block 256 = 8 waves; wave w owns 16 query
// rows. 32-key K/V tiles staged in LDS (K via async copies); online softmax.
// q:[S][H*192] bf16, kv:[S][H*256] bf16 (nope|v), kpe:[S][64] bf16,
// o:[S][H*128] bf16.
// ---------------------------------------------------------------------------
__global__ __launch_bounds__(256) void mla_attn_kernel(
    const __bf16* __restrict__ q, const __bf16* __restrict__ kv,
    const __bf16* __restrict__ kpe, __bf16* __restrict__ o, int S) {
    const int h = blockIdx.y, qb = blockIdx.x;
    const int tid = threadIdx.x, lane = tid & 31, wave = tid >> 5;
    const int qrow0 = qb * 128 + wave * 16;

    __shared__ __bf16 Ks[32][200];    // [key][d 0..191]; 400B rows (16B mult)
    __shared__ __bf16 Vs[128][40];    // [d][key] transposed
    __shared__ __bf16 Ps[8][16][40];  // per-wave P scratch (16x32)

    // Q fragments (resident for the whole kernel)
    bf16x16 qf[6];
    {
        int r  = qrow0 + (lane & 15);
        int kb = (lane < 16) ? 0 : 8;
        const __bf16* qp = q + (size_t)r * (NHEAD * Q_HEAD) + h * Q_HEAD;
#pragma unroll
        for (int f = 0; f < 6; ++f)
            qf[f] = cat16(ld8(qp + f * 32 + kb), ld8(qp + f * 32 + kb + 16));
    }

    f32x8 oacc[8];
#pragma unroll
    for (int i = 0; i < 8; ++i) oacc[i] = zero8();
    float mrow[8], srow[8];
#pragma unroll
    for (int i = 0; i < 8; ++i) { mrow[i] = -1e30f; srow[i] = 0.f; }

    const float scale = 0.0721687836487032f;  // 192^-0.5
    const int nkb = qb * 4 + 4;               // 32-key blocks (causal limit)
    const int rhalf = 8 * (lane >> 4);
    const int cl = lane & 15;

    for (int kb = 0; kb < nkb; ++kb) {
        const int key0 = kb * 32;
        // ---- stage K tile via async 16B copies ----
#pragma unroll
        for (int i = 0; i < 3; ++i) {
            int cg = tid + i * 256;  // 768 chunks of 8 bf16
            int key = cg / 24, dc = (cg % 24) * 8;
            const __bf16* src =
                (dc < 128)
                    ? kv + (size_t)(key0 + key) * (NHEAD * 256) + h * 256 + dc
                    : kpe + (size_t)(key0 + key) * 64 + (dc - 128);
            cp16(&Ks[key][dc], src);
        }
        // ---- stage V tile transposed (manual) ----
#pragma unroll
        for (int i = 0; i < 2; ++i) {
            int cg = tid + i * 256;  // 512 chunks of 8 bf16
            int key = cg >> 4, dc = (cg & 15) * 8;
            bf16x8 v = ld8(kv + (size_t)(key0 + key) * (NHEAD * 256) + h * 256 + 128 + dc);
#pragma unroll
            for (int j = 0; j < 8; ++j) Vs[dc + j][key] = v[j];
        }
        async_wait0();
        __syncthreads();

        if (key0 <= qrow0 + 15) {  // wave has at least one unmasked key here
            // ---- scores: two 16x16 tiles, 2 parallel accumulator chains ----
            f32x8 sc[2];
#pragma unroll
            for (int t = 0; t < 2; ++t) {
                f32x8 s0 = zero8(), s1 = zero8();
                int keyc = t * 16 + cl;
                int kb2  = (lane < 16) ? 0 : 8;
#pragma unroll
                for (int f = 0; f < 6; f += 2) {
                    bf16x16 b0 = cat16(ld8(&Ks[keyc][f * 32 + kb2]),
                                       ld8(&Ks[keyc][f * 32 + kb2 + 16]));
                    bf16x16 b1 = cat16(ld8(&Ks[keyc][(f + 1) * 32 + kb2]),
                                       ld8(&Ks[keyc][(f + 1) * 32 + kb2 + 16]));
                    s0 = wmma_bf16(qf[f], b0, s0);
                    s1 = wmma_bf16(qf[f + 1], b1, s1);
                }
                sc[t] = s0 + s1;
            }
            // ---- scale + causal mask ----
#pragma unroll
            for (int t = 0; t < 2; ++t)
#pragma unroll
                for (int i = 0; i < 8; ++i) {
                    int keyg = key0 + t * 16 + cl;
                    int rowg = qrow0 + i + rhalf;
                    float v = sc[t][i] * scale;
                    sc[t][i] = (keyg > rowg) ? -1e30f : v;
                }
            // ---- online softmax (per-row, 16-lane half-wave reductions) ----
            float corr[8];
#pragma unroll
            for (int i = 0; i < 8; ++i) {
                float m = fmaxf(sc[0][i], sc[1][i]);
#pragma unroll
                for (int msk = 8; msk >= 1; msk >>= 1)
                    m = fmaxf(m, __shfl_xor(m, msk, 32));
                float mn = fmaxf(mrow[i], m);
                corr[i] = __expf(mrow[i] - mn);
                mrow[i] = mn;
                float p0 = __expf(sc[0][i] - mn);
                float p1 = __expf(sc[1][i] - mn);
                sc[0][i] = p0; sc[1][i] = p1;
                float rs = p0 + p1;
#pragma unroll
                for (int msk = 8; msk >= 1; msk >>= 1)
                    rs += __shfl_xor(rs, msk, 32);
                srow[i] = srow[i] * corr[i] + rs;
            }
#pragma unroll
            for (int nc = 0; nc < 8; ++nc)
#pragma unroll
                for (int i = 0; i < 8; ++i) oacc[nc][i] *= corr[i];

            // ---- P (C layout) -> LDS -> A fragment (same-wave RAW) ----
#pragma unroll
            for (int t = 0; t < 2; ++t)
#pragma unroll
                for (int i = 0; i < 8; ++i)
                    Ps[wave][i + rhalf][t * 16 + cl] = (__bf16)sc[t][i];
            lds_raw_fence();
            bf16x16 pa;
            {
                int pr = lane & 15;
                int pk = (lane < 16) ? 0 : 8;
                pa = cat16(ld8(&Ps[wave][pr][pk]), ld8(&Ps[wave][pr][pk + 16]));
            }
            // ---- O += P @ V ----
#pragma unroll
            for (int nc = 0; nc < 8; ++nc) {
                int d  = nc * 16 + cl;
                int kk = (lane < 16) ? 0 : 8;
                bf16x16 vb = cat16(ld8(&Vs[d][kk]), ld8(&Vs[d][kk + 16]));
                oacc[nc] = wmma_bf16(pa, vb, oacc[nc]);
            }
        }
        __syncthreads();
    }

    // ---- epilogue: O /= rowsum, write bf16 ----
#pragma unroll
    for (int nc = 0; nc < 8; ++nc)
#pragma unroll
        for (int i = 0; i < 8; ++i) {
            int rowg = qrow0 + i + rhalf;
            float v = oacc[nc][i] / srow[i];
            o[(size_t)rowg * (NHEAD * V_DIM) + h * V_DIM + nc * 16 + cl] = (__bf16)v;
        }
}

// ---------------------------------------------------------------------------
// Host launcher
// ---------------------------------------------------------------------------
static inline int cdiv(int a, int b) { return (a + b - 1) / b; }

extern "C" void kernel_launch(void* const* d_in, const int* in_sizes, int n_in,
                              void* d_out, int out_size, void* d_ws, size_t ws_size,
                              hipStream_t stream) {
    const float* hidden     = (const float*)d_in[0];
    const float* w_qkv_a    = (const float*)d_in[1];
    const float* q_a_gamma  = (const float*)d_in[2];
    const float* w_q_b      = (const float*)d_in[3];
    const float* kv_a_gamma = (const float*)d_in[4];
    const float* w_kv_b     = (const float*)d_in[5];
    const float* w_o        = (const float*)d_in[6];
    const float* b_o        = (const float*)d_in[7];
    float* out = (float*)d_out;

    const int S = in_sizes[0] / HIDDEN;  // 2048

    char* ws = (char*)d_ws;
    size_t off = 0;
    auto alloc = [&](size_t bytes) -> void* {
        void* p = ws + off;
        off += (bytes + 255) & ~(size_t)255;
        return p;
    };
    float*  fused = (float*)alloc((size_t)S * NA * sizeof(float));
    __bf16* qcn   = (__bf16*)alloc((size_t)S * Q_LORA * 2);
    __bf16* ckvn  = (__bf16*)alloc((size_t)S * KV_LORA * 2);
    __bf16* kpe   = (__bf16*)alloc((size_t)S * ROPE * 2);
    __bf16* qbuf  = (__bf16*)alloc((size_t)S * NHEAD * Q_HEAD * 2);
    __bf16* kvbuf = (__bf16*)alloc((size_t)S * NHEAD * 256 * 2);
    __bf16* attn  = (__bf16*)alloc((size_t)S * NHEAD * V_DIM * 2);
    (void)ws_size; (void)n_in; (void)out_size;

    // 1) fused = hidden @ w_qkv_a   (f32 A, f32 C)
    gemm_bf16_kernel<true, true, false>
        <<<dim3(cdiv(NA, 128), S / 128), 256, 0, stream>>>(
            hidden, w_qkv_a, fused, nullptr, S, NA, HIDDEN);

    // 2) rmsnorm q_c / ckv, convert k_pe
    rmsnorm_kernel<Q_LORA><<<S, 256, 0, stream>>>(fused, NA, q_a_gamma, qcn);
    rmsnorm_kernel<KV_LORA><<<S, 256, 0, stream>>>(fused + Q_LORA, NA, kv_a_gamma, ckvn);
    kpe_cvt_kernel<<<cdiv(S * ROPE, 256), 256, 0, stream>>>(fused, kpe, S * ROPE);

    // 3) q = qcn @ w_q_b (bf16 C), kv = ckvn @ w_kv_b (bf16 C)
    gemm_bf16_kernel<false, false, false>
        <<<dim3(cdiv(NHEAD * Q_HEAD, 128), S / 128), 256, 0, stream>>>(
            qcn, w_q_b, qbuf, nullptr, S, NHEAD * Q_HEAD, Q_LORA);
    gemm_bf16_kernel<false, false, false>
        <<<dim3(cdiv(NHEAD * 256, 128), S / 128), 256, 0, stream>>>(
            ckvn, w_kv_b, kvbuf, nullptr, S, NHEAD * 256, KV_LORA);

    // 4) flash attention
    mla_attn_kernel<<<dim3(S / 128, NHEAD), 256, 0, stream>>>(qbuf, kvbuf, kpe, attn, S);

    // 5) out = attn @ w_o + b_o  (f32 C, bias)
    gemm_bf16_kernel<false, true, true>
        <<<dim3(cdiv(HIDDEN, 128), S / 128), 256, 0, stream>>>(
            attn, w_o, out, b_o, S, HIDDEN, NHEAD * V_DIM);
}
